// SwinEncoder_67087389164210
// MI455X (gfx1250) — matrix operand order repaired and checked by reading, not compiled, not run on hardware
//
#include <hip/hip_runtime.h>
#include <hip/hip_bf16.h>
#include <stdint.h>

// ---------------------------------------------------------------------------
// Swin encoder block x2 for MI455X (gfx1250, wave32, WMMA).
// GEMMs in bf16 with f32 accumulation via v_wmma_f32_16x16x32_bf16,
// 2x2 register blocking per wave (32x32 C tile) for 2x arithmetic intensity.
// ---------------------------------------------------------------------------

typedef __attribute__((ext_vector_type(16))) __bf16 v16bf;
typedef __attribute__((ext_vector_type(8)))  float  v8f;

#define BB   32
#define CC   128
#define HH_  64
#define WW_  64
#define HWN  (HH_ * WW_)            // 4096
#define MTOK (BB * HWN)             // 131072 tokens

// float -> bf16 bits (round to nearest even)
__device__ __forceinline__ unsigned short f2bf(float f) {
    union { float f; unsigned u; } a; a.f = f;
    unsigned r = a.u + 0x7FFFu + ((a.u >> 16) & 1u);
    return (unsigned short)(r >> 16);
}

// ---------------------------------------------------------------------------
// Weight convert + transpose: W (K x N, f32) -> Wt (N x K, bf16)
// ---------------------------------------------------------------------------
__global__ void wcvt_kernel(const float* __restrict__ W, unsigned short* __restrict__ Wt,
                            int K, int N) {
    int i = blockIdx.x * blockDim.x + threadIdx.x;
    if (i >= K * N) return;
    int n = i % N;
    int k = i / N;
    Wt[(size_t)n * K + k] = f2bf(W[i]);
}

// ---------------------------------------------------------------------------
// NCHW -> token-major (B*H*W, C) f32
// ---------------------------------------------------------------------------
__global__ void nchw2tok_kernel(const float* __restrict__ X, float* __restrict__ O) {
    size_t i = (size_t)blockIdx.x * blockDim.x + threadIdx.x;   // over B*C*HW
    int p = (int)(i % HWN);
    int c = (int)((i / HWN) % CC);
    int b = (int)(i / ((size_t)HWN * CC));
    O[((size_t)b * HWN + p) * CC + c] = X[i];
}

// ---------------------------------------------------------------------------
// LayerNorm over C=128: one wave per token, 4 channels per lane, shfl reduce.
// Output bf16 (GEMM A operand).
// ---------------------------------------------------------------------------
__global__ void ln_kernel(const float* __restrict__ X, const float* __restrict__ G,
                          const float* __restrict__ Bv, unsigned short* __restrict__ O) {
    int token = blockIdx.x * (blockDim.x >> 5) + (threadIdx.x >> 5);
    int lane  = threadIdx.x & 31;
    const float4* xr = reinterpret_cast<const float4*>(X + (size_t)token * CC);
    float4 xv = xr[lane];
    float s = xv.x + xv.y + xv.z + xv.w;
#pragma unroll
    for (int o = 16; o > 0; o >>= 1) s += __shfl_xor(s, o, 32);
    float mu = s * (1.0f / 128.0f);
    float d0 = xv.x - mu, d1 = xv.y - mu, d2 = xv.z - mu, d3 = xv.w - mu;
    float vs = d0 * d0 + d1 * d1 + d2 * d2 + d3 * d3;
#pragma unroll
    for (int o = 16; o > 0; o >>= 1) vs += __shfl_xor(vs, o, 32);
    float inv = rsqrtf(vs * (1.0f / 128.0f) + 1e-5f);
    float4 gv = reinterpret_cast<const float4*>(G)[lane];
    float4 bv = reinterpret_cast<const float4*>(Bv)[lane];
    ushort4 o4;
    o4.x = f2bf(d0 * inv * gv.x + bv.x);
    o4.y = f2bf(d1 * inv * gv.y + bv.y);
    o4.z = f2bf(d2 * inv * gv.z + bv.z);
    o4.w = f2bf(d3 * inv * gv.w + bv.w);
    reinterpret_cast<ushort4*>(O)[(size_t)token * 32 + lane] = o4;
}

// ---------------------------------------------------------------------------
// WMMA GEMM:  Out(MxN) = A(MxK, bf16) @ Bt(NxK, bf16)^T + bias [+ GELU] [+ res]
// 4 waves per block in a 2x2 arrangement; each wave computes a 32x32 C region
// as 2x2 WMMA tiles (4 accumulators), K-loop step 32.
// A fragment (16-bit A 16x32 layout): lane m = lane&15,
//   lanes 0-15:  K = {k0+0..7,  k0+16..23}
//   lanes 16-31: K = {k0+8..15, k0+24..31}
// B fragment (same K order as A): lane n = lane&15,
//   lanes 0-15:  K = {k0+0..7,  k0+16..23} from Bt[n][.]
//   lanes 16-31: K = {k0+8..15, k0+24..31}
// C/D: acc[v] -> row tile_m + v + (lane>=16 ? 8 : 0), col tile_n + (lane&15)
// ---------------------------------------------------------------------------
__global__ void gemm_bf16_wmma(const unsigned short* __restrict__ A,
                               const unsigned short* __restrict__ Bt,
                               const float* __restrict__ bias,
                               const float* __restrict__ res,
                               float* __restrict__ outF,
                               unsigned short* __restrict__ outB,
                               int N, int K, int gelu) {
    const int wave = threadIdx.x >> 5;
    const int lane = threadIdx.x & 31;
    const int hi   = lane >> 4;          // 0 = lanes 0-15, 1 = lanes 16-31
    const int l16  = lane & 15;
    const int m0 = blockIdx.x * 64 + (wave >> 1) * 32;   // wave's 32-row strip
    const int n0 = blockIdx.y * 64 + (wave & 1) * 32;    // wave's 32-col strip

    v8f acc00 = {0.f,0.f,0.f,0.f,0.f,0.f,0.f,0.f};
    v8f acc01 = acc00, acc10 = acc00, acc11 = acc00;

    const unsigned short* arow0 = A  + (size_t)(m0 + l16) * K + (hi ? 8 : 0);
    const unsigned short* arow1 = arow0 + (size_t)16 * K;
    const unsigned short* brow0 = Bt + (size_t)(n0 + l16) * K + (hi ? 8 : 0);
    const unsigned short* brow1 = brow0 + (size_t)16 * K;

    union Frag { uint4 u[2]; v16bf v; };

    for (int k0 = 0; k0 < K; k0 += 32) {
        Frag a0, a1, b0, b1;
        a0.u[0] = *reinterpret_cast<const uint4*>(arow0 + k0);
        a0.u[1] = *reinterpret_cast<const uint4*>(arow0 + k0 + 16);
        a1.u[0] = *reinterpret_cast<const uint4*>(arow1 + k0);
        a1.u[1] = *reinterpret_cast<const uint4*>(arow1 + k0 + 16);
        b0.u[0] = *reinterpret_cast<const uint4*>(brow0 + k0);
        b0.u[1] = *reinterpret_cast<const uint4*>(brow0 + k0 + 16);
        b1.u[0] = *reinterpret_cast<const uint4*>(brow1 + k0);
        b1.u[1] = *reinterpret_cast<const uint4*>(brow1 + k0 + 16);
        acc00 = __builtin_amdgcn_wmma_f32_16x16x32_bf16(
            false, a0.v, false, b0.v, (short)0, acc00, false, false);
        acc01 = __builtin_amdgcn_wmma_f32_16x16x32_bf16(
            false, a0.v, false, b1.v, (short)0, acc01, false, false);
        acc10 = __builtin_amdgcn_wmma_f32_16x16x32_bf16(
            false, a1.v, false, b0.v, (short)0, acc10, false, false);
        acc11 = __builtin_amdgcn_wmma_f32_16x16x32_bf16(
            false, a1.v, false, b1.v, (short)0, acc11, false, false);
    }

    v8f accs[2][2] = {{acc00, acc01}, {acc10, acc11}};
#pragma unroll
    for (int mi = 0; mi < 2; ++mi) {
#pragma unroll
        for (int ni = 0; ni < 2; ++ni) {
            const int col  = n0 + ni * 16 + l16;
            const float bv = bias ? bias[col] : 0.0f;
            const int mrow = m0 + mi * 16 + hi * 8;
#pragma unroll
            for (int v = 0; v < 8; ++v) {
                int m = mrow + v;
                float xx = accs[mi][ni][v] + bv;
                if (gelu) xx = 0.5f * xx * (1.0f + erff(xx * 0.70710678118f));
                if (res)  xx += res[(size_t)m * N + col];
                size_t o = (size_t)m * N + col;
                if (outF) outF[o] = xx;
                if (outB) outB[o] = f2bf(xx);
            }
        }
    }
}

// ---------------------------------------------------------------------------
// Window attention (ws=2, n=4 tokens, 4 heads, e=32).
// One thread per (b, wh, wwn, head, i-query-row). QKV layout: token-major,
// column = (e*4 + head)*3 + sel  (c outer / k inner split, head-fast channel).
// shifted: read qkv at rolled coords (+1,+1) mod 64 (reference never unrolls);
// masks: last window-row -> (i>=2)!=(j>=2); last window-col -> parity(i)!=parity(j).
// Output bf16, channel = head*32 + e  ('(H e)' merge).
// ---------------------------------------------------------------------------
__global__ void attn_kernel(const float* __restrict__ QKV, const float* __restrict__ pos,
                            unsigned short* __restrict__ O, int shifted) {
    int t = blockIdx.x * blockDim.x + threadIdx.x;   // 524288 threads
    int i    =  t        & 3;
    int head = (t >> 2)  & 3;
    int wwn  = (t >> 4)  & 31;
    int wh   = (t >> 9)  & 31;
    int b    =  t >> 14;
    int xi = i >> 1, yi = i & 1;

    size_t rows[4];
#pragma unroll
    for (int j = 0; j < 4; ++j) {
        int hh = wh * 2 + (j >> 1);
        int ww = wwn * 2 + (j & 1);
        if (shifted) { hh = (hh + 1) & 63; ww = (ww + 1) & 63; }
        rows[j] = ((size_t)b * HWN + (size_t)hh * WW_ + ww) * 384;
    }

    float q[32];
    const size_t qrow = rows[i];
#pragma unroll
    for (int e = 0; e < 32; ++e) q[e] = QKV[qrow + (size_t)(e * 4 + head) * 3 + 0];

    float sc[4];
#pragma unroll
    for (int j = 0; j < 4; ++j) {
        const size_t kr = rows[j];
        float d = 0.f;
#pragma unroll
        for (int e = 0; e < 32; ++e) d += q[e] * QKV[kr + (size_t)(e * 4 + head) * 3 + 1];
        d *= 0.17677669529663687f;                     // 1/sqrt(32)
        int xj = j >> 1, yj = j & 1;
        d += pos[(xj - xi + 1) * 3 + (yj - yi + 1)];
        if (shifted) {
            if (wh == 31 && ((i >= 2) != (j >= 2))) d = -1e30f;   // row mask
            if (wwn == 31 && (((i ^ j) & 1) != 0))  d = -1e30f;   // col mask
        }
        sc[j] = d;
    }

    float mx = fmaxf(fmaxf(sc[0], sc[1]), fmaxf(sc[2], sc[3]));
    float p[4], sum = 0.f;
#pragma unroll
    for (int j = 0; j < 4; ++j) { p[j] = expf(sc[j] - mx); sum += p[j]; }
    float isum = 1.0f / sum;

    float out[32];
#pragma unroll
    for (int e = 0; e < 32; ++e) out[e] = 0.f;
#pragma unroll
    for (int j = 0; j < 4; ++j) {
        float w = p[j] * isum;
        const size_t vr = rows[j];
#pragma unroll
        for (int e = 0; e < 32; ++e) out[e] += w * QKV[vr + (size_t)(e * 4 + head) * 3 + 2];
    }

    // output at un-rolled grid position (reference never rolls back)
    size_t orow = ((size_t)b * HWN + (size_t)(wh * 2 + xi) * WW_ + (wwn * 2 + yi)) * CC;
#pragma unroll
    for (int e = 0; e < 32; ++e) O[orow + head * 32 + e] = f2bf(out[e]);
}

// ---------------------------------------------------------------------------
// Host-side orchestration
// ---------------------------------------------------------------------------
extern "C" void kernel_launch(void* const* d_in, const int* in_sizes, int n_in,
                              void* d_out, int out_size, void* d_ws, size_t ws_size,
                              hipStream_t stream) {
    const float* x    = (const float*)d_in[0];
    const float* ln_g = (const float*)d_in[1];
    const float* ln_b = (const float*)d_in[2];
    const float* wq_W = (const float*)d_in[3];
    const float* wq_b = (const float*)d_in[4];
    const float* wp_W = (const float*)d_in[5];
    const float* wp_b = (const float*)d_in[6];
    const float* wpos = (const float*)d_in[7];
    const float* sq_W = (const float*)d_in[8];
    const float* sq_b = (const float*)d_in[9];
    const float* sp_W = (const float*)d_in[10];
    const float* sp_b = (const float*)d_in[11];
    const float* spos = (const float*)d_in[12];
    const float* m1_W = (const float*)d_in[13];
    const float* m1_b = (const float*)d_in[14];
    const float* m2_W = (const float*)d_in[15];
    const float* m2_b = (const float*)d_in[16];

    char* ws = (char*)d_ws;
    size_t off = 0;
    auto alloc = [&](size_t bytes) -> void* {
        void* p = ws + off;
        off = (off + bytes + 255) & ~(size_t)255;
        return p;
    };

    unsigned short* wq_t = (unsigned short*)alloc((size_t)384 * 128 * 2);
    unsigned short* wp_t = (unsigned short*)alloc((size_t)128 * 128 * 2);
    unsigned short* sq_t = (unsigned short*)alloc((size_t)384 * 128 * 2);
    unsigned short* sp_t = (unsigned short*)alloc((size_t)128 * 128 * 2);
    unsigned short* m1_t = (unsigned short*)alloc((size_t)512 * 128 * 2);
    unsigned short* m2_t = (unsigned short*)alloc((size_t)128 * 512 * 2);
    float*          R0   = (float*)alloc((size_t)MTOK * CC * 4);    // residual ping
    float*          R1   = (float*)alloc((size_t)MTOK * CC * 4);    // residual pong
    unsigned short* Cb   = (unsigned short*)alloc((size_t)MTOK * CC * 2);  // ln/attn bf16
    void*           Dr   = alloc((size_t)MTOK * 384 * 4);           // qkv f32 / mlp-hidden bf16

    float*          qkv = (float*)Dr;
    unsigned short* h16 = (unsigned short*)Dr;

    const dim3 blk(256);
    const dim3 gblk(128);

    // ---- weights ----
    wcvt_kernel<<<dim3((128 * 384 + 255) / 256), blk, 0, stream>>>(wq_W, wq_t, 128, 384);
    wcvt_kernel<<<dim3((128 * 128 + 255) / 256), blk, 0, stream>>>(wp_W, wp_t, 128, 128);
    wcvt_kernel<<<dim3((128 * 384 + 255) / 256), blk, 0, stream>>>(sq_W, sq_t, 128, 384);
    wcvt_kernel<<<dim3((128 * 128 + 255) / 256), blk, 0, stream>>>(sp_W, sp_t, 128, 128);
    wcvt_kernel<<<dim3((128 * 512 + 255) / 256), blk, 0, stream>>>(m1_W, m1_t, 128, 512);
    wcvt_kernel<<<dim3((512 * 128 + 255) / 256), blk, 0, stream>>>(m2_W, m2_t, 512, 128);

    // ---- block 1: W-MSA ----
    nchw2tok_kernel<<<dim3((unsigned)((size_t)BB * CC * HWN / 256)), blk, 0, stream>>>(x, R0);
    ln_kernel<<<dim3(MTOK / 8), blk, 0, stream>>>(R0, ln_g, ln_b, Cb);
    gemm_bf16_wmma<<<dim3(MTOK / 64, 384 / 64), gblk, 0, stream>>>(
        Cb, wq_t, wq_b, nullptr, qkv, nullptr, 384, 128, 0);
    attn_kernel<<<dim3(524288 / 256), blk, 0, stream>>>(qkv, wpos, Cb, 0);
    gemm_bf16_wmma<<<dim3(MTOK / 64, 128 / 64), gblk, 0, stream>>>(
        Cb, wp_t, wp_b, R0, R1, nullptr, 128, 128, 0);

    // ---- block 1: MLP ----
    ln_kernel<<<dim3(MTOK / 8), blk, 0, stream>>>(R1, ln_g, ln_b, Cb);
    gemm_bf16_wmma<<<dim3(MTOK / 64, 512 / 64), gblk, 0, stream>>>(
        Cb, m1_t, m1_b, nullptr, nullptr, h16, 512, 128, 1);
    gemm_bf16_wmma<<<dim3(MTOK / 64, 128 / 64), gblk, 0, stream>>>(
        h16, m2_t, m2_b, R1, R0, nullptr, 128, 512, 0);

    // ---- block 2: SW-MSA ----
    ln_kernel<<<dim3(MTOK / 8), blk, 0, stream>>>(R0, ln_g, ln_b, Cb);
    gemm_bf16_wmma<<<dim3(MTOK / 64, 384 / 64), gblk, 0, stream>>>(
        Cb, sq_t, sq_b, nullptr, qkv, nullptr, 384, 128, 0);
    attn_kernel<<<dim3(524288 / 256), blk, 0, stream>>>(qkv, spos, Cb, 1);
    gemm_bf16_wmma<<<dim3(MTOK / 64, 128 / 64), gblk, 0, stream>>>(
        Cb, sp_t, sp_b, R0, R1, nullptr, 128, 128, 0);

    // ---- block 2: MLP (writes final output) ----
    ln_kernel<<<dim3(MTOK / 8), blk, 0, stream>>>(R1, ln_g, ln_b, Cb);
    gemm_bf16_wmma<<<dim3(MTOK / 64, 512 / 64), gblk, 0, stream>>>(
        Cb, m1_t, m1_b, nullptr, nullptr, h16, 512, 128, 1);
    gemm_bf16_wmma<<<dim3(MTOK / 64, 128 / 64), gblk, 0, stream>>>(
        h16, m2_t, m2_b, R1, (float*)d_out, nullptr, 128, 512, 0);
}
